// MSAColGlobalAttention_37417755083156
// MI455X (gfx1250) — matrix operand order repaired and checked by reading, not compiled, not run on hardware
//
#include <hip/hip_runtime.h>
#include <hip/hip_bf16.h>
#include <math.h>

typedef __attribute__((ext_vector_type(16))) _Float16 v16h;
typedef __attribute__((ext_vector_type(8)))  _Float16 v8h;
typedef __attribute__((ext_vector_type(8)))  float    v8f;

#define NN   512     // N (rows to attend over)
#define LL   1024    // L (columns)
#define DD   64      // D_MSA
#define HH   8       // heads
#define DH   8       // head dim
#define LNE  1e-5f

// ---------------- workspace layout ----------------
// xsum : [L][64] f32                 65536 f32
// outw : [L][64] f32                 65536 f32
// kws  : [L][N][8] f16               4 Mi f16
// vws  : [L][N][8] f16               4 Mi f16
// bfKV : [kc=2][lane=32][16] f16     1024 f16   (Wk|Wv fragments)
// bfG  : [ct=4][kc=2][lane=32][16]   4096 f16   (Wg fragments)
// bfO  : [ct=4][kc=2][lane=32][16]   4096 f16   (Wo fragments)

__global__ void mcga_zero(float* xsum) {
    int i = blockIdx.x * blockDim.x + threadIdx.x;
    if (i < LL * DD) xsum[i] = 0.f;
}

// Precompute WMMA B-fragments (per-lane layout per ISA 7.12.2) in f16.
__global__ void mcga_prepw(const float* __restrict__ Wk,
                           const float* __restrict__ Wv,
                           const float* __restrict__ Wg,
                           const float* __restrict__ Wo,
                           _Float16* __restrict__ bfKV,
                           _Float16* __restrict__ bfG,
                           _Float16* __restrict__ bfO) {
    for (int idx = threadIdx.x; idx < 9216; idx += blockDim.x) {
        if (idx < 1024) {                       // KV: [kc][lane][i]
            const int j = idx;
            const int kc = j >> 9, lane = (j >> 4) & 31, i = j & 15;
            const int ncol = lane & 15;
            const int kk = kc * 32 + ((lane < 16) ? 0 : 8) + ((i < 8) ? i : i + 8);
            const float v = (ncol < 8) ? Wk[kk * 8 + ncol] : Wv[kk * 8 + (ncol - 8)];
            bfKV[j] = (_Float16)v;
        } else if (idx < 5120) {                // G: [ct][kc][lane][i]
            const int j = idx - 1024;
            const int ct = j >> 10, kc = (j >> 9) & 1, lane = (j >> 4) & 31, i = j & 15;
            const int ncol = lane & 15;
            const int kk = kc * 32 + ((lane < 16) ? 0 : 8) + ((i < 8) ? i : i + 8);
            bfG[j] = (_Float16)Wg[kk * 64 + ct * 16 + ncol];
        } else {                                // O: [ct][kc][lane][i]
            const int j = idx - 5120;
            const int ct = j >> 10, kc = (j >> 9) & 1, lane = (j >> 4) & 31, i = j & 15;
            const int ncol = lane & 15;
            const int kk = kc * 32 + ((lane < 16) ? 0 : 8) + ((i < 8) ? i : i + 8);
            bfO[j] = (_Float16)Wo[kk * 64 + ct * 16 + ncol];
        }
    }
}

// ---------------- Pass 1: LN + k/v projection (WMMA) + column sums ---
// tile = 16 n-rows x 4 l-cols = 64 rows, block = 256 threads (8 waves)
__global__ void mcga_pass1(const float* __restrict__ msa,
                           const float* __restrict__ ln_w,
                           const float* __restrict__ ln_b,
                           const _Float16* __restrict__ bfKV,
                           float* __restrict__ xsum,
                           _Float16* __restrict__ kws,
                           _Float16* __restrict__ vws) {
    __shared__ __align__(16) float    x32[64 * 64];
    __shared__ __align__(16) _Float16 x16[64 * 64];
    __shared__ __align__(16) _Float16 kvstage[64 * 16];

    const int tid = threadIdx.x;
    const int tl  = blockIdx.x & 255;        // 256 l-tiles
    const int tn  = blockIdx.x >> 8;         // 32 n-tiles
    const int l0  = tl * 4;
    const int n0  = tn * 16;

    // ---- LayerNorm: 4 threads/row, each loads 4x float4 (64B groups) --
    {
        const int r = tid >> 2, t = tid & 3;
        const int nI = n0 + (r >> 2), lI = l0 + (r & 3);
        const float4* src = (const float4*)(msa + ((size_t)nI * LL + lI) * DD);
        float4 vv[4]; float s = 0.f, s2 = 0.f;
        #pragma unroll
        for (int i = 0; i < 4; ++i) {
            vv[i] = src[t + 4 * i];
            s  += vv[i].x + vv[i].y + vv[i].z + vv[i].w;
            s2 += vv[i].x * vv[i].x + vv[i].y * vv[i].y
                + vv[i].z * vv[i].z + vv[i].w * vv[i].w;
        }
        s  += __shfl_xor(s, 1);  s  += __shfl_xor(s, 2);
        s2 += __shfl_xor(s2, 1); s2 += __shfl_xor(s2, 2);
        const float mu  = s * (1.f / 64.f);
        const float var = s2 * (1.f / 64.f) - mu * mu;
        const float rs  = rsqrtf(var + LNE);
        #pragma unroll
        for (int i = 0; i < 4; ++i) {
            const float* e = (const float*)&vv[i];
            #pragma unroll
            for (int j = 0; j < 4; ++j) {
                const int d = (t + 4 * i) * 4 + j;
                const float xn = (e[j] - mu) * rs * ln_w[d] + ln_b[d];
                x32[r * 64 + d] = xn;
                x16[r * 64 + d] = (_Float16)xn;
            }
        }
    }
    __syncthreads();

    // ---- partial column sums -> atomic add into xsum[l][d] ----
    {
        const int ll = tid >> 6;
        const int d  = tid & 63;
        float s = 0.f;
        #pragma unroll
        for (int j = 0; j < 16; ++j) s += x32[(4 * j + ll) * 64 + d];
        atomicAdd(&xsum[(l0 + ll) * DD + d], s);
    }

    // ---- k/v projection: X[64x64] @ [Wk|Wv][64x16] via WMMA f16 ----
    const int w    = tid >> 5;
    const int lane = tid & 31;
    if (w < 4) {
        const int m0   = w * 16;
        const int arow = m0 + (lane & 15);
        const int koff = (lane < 16) ? 0 : 8;
        const int ncol = lane & 15;
        v8f c = {};
        #pragma unroll
        for (int kc = 0; kc < 2; ++kc) {
            v16h a;
            const _Float16* xr = &x16[arow * 64 + kc * 32];
            #pragma unroll
            for (int i = 0; i < 8; ++i) {
                a[i]     = xr[koff + i];
                a[8 + i] = xr[16 + koff + i];
            }
            const v16h b = *(const v16h*)(bfKV + ((kc * 32 + lane) << 4));
            c = __builtin_amdgcn_wmma_f32_16x16x32_f16(false, a, false, b,
                                                       (short)0, c, false, false);
        }
        #pragma unroll
        for (int vi = 0; vi < 8; ++vi) {
            const int m = vi + ((lane >> 4) << 3);
            kvstage[(m0 + m) * 16 + ncol] = (_Float16)c[vi];
        }
    }
    __syncthreads();

    // ---- coalesced 16B row stores of k and v ----
    if (tid < 128) {
        const int r = tid >> 1, half = tid & 1;
        const int nI = n0 + (r >> 2), lI = l0 + (r & 3);
        const uint4 val = *(const uint4*)&kvstage[r * 16 + half * 8];
        _Float16* dst = (half == 0)
            ? (kws + (size_t)lI * (NN * 8) + nI * 8)
            : (vws + (size_t)lI * (NN * 8) + nI * 8);
        *(uint4*)dst = val;
    }
}

// ---------------- Pass 2: q + softmax attention over n, per column l --
__global__ void mcga_pass2(const float* __restrict__ xsum,
                           const float* __restrict__ Wq,
                           const _Float16* __restrict__ kws,
                           const _Float16* __restrict__ vws,
                           float* __restrict__ outw) {
    __shared__ __align__(16) float    xb[64];
    __shared__ __align__(16) float    qs[64];
    __shared__ __align__(16) float    lg[HH * NN];   // 16 KB logits / attn
    __shared__ __align__(16) _Float16 ksh[NN * 8];   // 8 KB, [n][d]
    __shared__ __align__(16) _Float16 vsh[8 * NN];   // 8 KB, [d][n] (transposed)
    __shared__ float hs[HH];
    const int l   = blockIdx.x;
    const int tid = threadIdx.x;

    // ---- coalesced staging of k (as-is) and v (transposed) ----
    {
        const uint4* ksrc = (const uint4*)(kws + (size_t)l * (NN * 8));
        const uint4* vsrc = (const uint4*)(vws + (size_t)l * (NN * 8));
        #pragma unroll
        for (int it = 0; it < 2; ++it) {
            const int n = tid + it * 256;            // row index
            ((uint4*)ksh)[n] = ksrc[n];
            union { uint4 u; _Float16 h[8]; } val;
            val.u = vsrc[n];
            #pragma unroll
            for (int d = 0; d < 8; ++d) vsh[d * NN + n] = val.h[d];
        }
    }
    if (tid < 64) xb[tid] = xsum[l * DD + tid] * (1.f / (float)NN);
    __syncthreads();
    if (tid < 64) {
        float acc = 0.f;
        #pragma unroll
        for (int d = 0; d < 64; ++d) acc += xb[d] * Wq[d * 64 + tid];
        qs[tid] = acc * 0.35355339059327373f;   // 1/sqrt(8)
    }
    __syncthreads();

    // logits[h][n] = q[h] . k[n]  (k row = one ds_load_b128)
    #pragma unroll 4
    for (int it = 0; it < 16; ++it) {
        const int idx = it * 256 + tid;
        const int n = idx >> 3, h = idx & 7;
        const v8h kv = *(const v8h*)(ksh + n * 8);
        float acc = 0.f;
        #pragma unroll
        for (int d = 0; d < DH; ++d) acc += qs[h * DH + d] * (float)kv[d];
        lg[h * NN + n] = acc;
    }
    __syncthreads();

    // softmax over n: one wave per head
    {
        const int w = tid >> 5, lane = tid & 31;
        float m = -1e30f;
        for (int n = lane; n < NN; n += 32) m = fmaxf(m, lg[w * NN + n]);
        #pragma unroll
        for (int msk = 16; msk >= 1; msk >>= 1) m = fmaxf(m, __shfl_xor(m, msk));
        float s = 0.f;
        for (int n = lane; n < NN; n += 32) {
            const float e = expf(lg[w * NN + n] - m);
            lg[w * NN + n] = e; s += e;
        }
        #pragma unroll
        for (int msk = 16; msk >= 1; msk >>= 1) s += __shfl_xor(s, msk);
        if (lane == 0) hs[w] = s;
    }
    __syncthreads();

    // out[l][h][dh] = (attn . v) / denom  -- all wide LDS reads
    if (tid < 64) {
        const int h = tid >> 3, d = tid & 7;
        float acc = 0.f;
        #pragma unroll 4
        for (int n = 0; n < NN; n += 8) {
            const v8h    vv = *(const v8h*)(vsh + d * NN + n);
            const float4 a0 = *(const float4*)(lg + h * NN + n);
            const float4 a1 = *(const float4*)(lg + h * NN + n + 4);
            acc += a0.x * (float)vv[0] + a0.y * (float)vv[1]
                 + a0.z * (float)vv[2] + a0.w * (float)vv[3]
                 + a1.x * (float)vv[4] + a1.y * (float)vv[5]
                 + a1.z * (float)vv[6] + a1.w * (float)vv[7];
        }
        outw[l * DD + tid] = acc / hs[h];
    }
}

// ---------------- Pass 3: re-LN + gate GEMM + output GEMM (WMMA) -----
// tile = 64 contiguous rows (same n, 64 consecutive l), block = 256 thr
__global__ void mcga_pass3(const float* __restrict__ msa,
                           const float* __restrict__ ln_w,
                           const float* __restrict__ ln_b,
                           const _Float16* __restrict__ bfG,
                           const float* __restrict__ bg,
                           const _Float16* __restrict__ bfO,
                           const float* __restrict__ bo,
                           const float* __restrict__ outw,
                           float* __restrict__ out) {
    __shared__ __align__(16) unsigned char smemraw[64 * 64 * 4];  // 16 KB
    _Float16* x16  = (_Float16*)smemraw;             // [0, 8K)
    _Float16* y16  = (_Float16*)(smemraw + 8192);    // [8K, 16K)
    float*    stage = (float*)smemraw;               // overlay, 16 KB

    const int tid  = threadIdx.x;
    const size_t rho0 = (size_t)blockIdx.x * 64;
    const int l0 = (int)(rho0 & (LL - 1));

    // ---- LayerNorm (contiguous, float4 loads) ----
    {
        const int r = tid >> 2, t = tid & 3;
        const float4* src = (const float4*)(msa + (rho0 + r) * DD);
        float4 vv[4]; float s = 0.f, s2 = 0.f;
        #pragma unroll
        for (int i = 0; i < 4; ++i) {
            vv[i] = src[t + 4 * i];
            s  += vv[i].x + vv[i].y + vv[i].z + vv[i].w;
            s2 += vv[i].x * vv[i].x + vv[i].y * vv[i].y
                + vv[i].z * vv[i].z + vv[i].w * vv[i].w;
        }
        s  += __shfl_xor(s, 1);  s  += __shfl_xor(s, 2);
        s2 += __shfl_xor(s2, 1); s2 += __shfl_xor(s2, 2);
        const float mu  = s * (1.f / 64.f);
        const float var = s2 * (1.f / 64.f) - mu * mu;
        const float rs  = rsqrtf(var + LNE);
        #pragma unroll
        for (int i = 0; i < 4; ++i) {
            const float* e = (const float*)&vv[i];
            #pragma unroll
            for (int j = 0; j < 4; ++j) {
                const int d = (t + 4 * i) * 4 + j;
                x16[r * 64 + d] = (_Float16)((e[j] - mu) * rs * ln_w[d] + ln_b[d]);
            }
        }
    }
    __syncthreads();

    const int w    = tid >> 5;
    const int lane = tid & 31;
    const int koff = (lane < 16) ? 0 : 8;
    const int ncol = lane & 15;

    // ---- GEMM1: gate = sigmoid(X @ Wg + bg); y = gate * out[l] ----
    #pragma unroll
    for (int s = 0; s < 2; ++s) {
        const int p  = w * 2 + s;
        const int rt = p >> 2, ct = p & 3;
        const int arow = rt * 16 + (lane & 15);
        v8f c = {};
        #pragma unroll
        for (int kc = 0; kc < 2; ++kc) {
            v16h a;
            const _Float16* xr = &x16[arow * 64 + kc * 32];
            #pragma unroll
            for (int i = 0; i < 8; ++i) {
                a[i]     = xr[koff + i];
                a[8 + i] = xr[16 + koff + i];
            }
            const v16h b = *(const v16h*)(bfG + (((ct * 2 + kc) * 32 + lane) << 4));
            c = __builtin_amdgcn_wmma_f32_16x16x32_f16(false, a, false, b,
                                                       (short)0, c, false, false);
        }
        #pragma unroll
        for (int vi = 0; vi < 8; ++vi) {
            const int m   = vi + ((lane >> 4) << 3);
            const int row = rt * 16 + m;
            const int col = ct * 16 + ncol;
            float g = c[vi] + bg[col];
            g = 1.f / (1.f + expf(-g));
            y16[row * 64 + col] = (_Float16)(g * outw[(l0 + row) * DD + col]);
        }
    }
    __syncthreads();

    // ---- GEMM2: out = Y @ Wo + bo (both accumulators kept in regs) ----
    v8f cs[2];
    #pragma unroll
    for (int s = 0; s < 2; ++s) {
        const int p  = w * 2 + s;
        const int rt = p >> 2, ct = p & 3;
        const int arow = rt * 16 + (lane & 15);
        v8f c = {};
        #pragma unroll
        for (int kc = 0; kc < 2; ++kc) {
            v16h a;
            const _Float16* yr = &y16[arow * 64 + kc * 32];
            #pragma unroll
            for (int i = 0; i < 8; ++i) {
                a[i]     = yr[koff + i];
                a[8 + i] = yr[16 + koff + i];
            }
            const v16h b = *(const v16h*)(bfO + (((ct * 2 + kc) * 32 + lane) << 4));
            c = __builtin_amdgcn_wmma_f32_16x16x32_f16(false, a, false, b,
                                                       (short)0, c, false, false);
        }
        cs[s] = c;
    }
    __syncthreads();                 // all reads of y16 done; reuse as f32 stage

    #pragma unroll
    for (int s = 0; s < 2; ++s) {
        const int p  = w * 2 + s;
        const int rt = p >> 2, ct = p & 3;
        #pragma unroll
        for (int vi = 0; vi < 8; ++vi) {
            const int m   = vi + ((lane >> 4) << 3);
            const int row = rt * 16 + m;
            const int col = ct * 16 + ncol;
            stage[row * 64 + col] = cs[s][vi] + bo[col];
        }
    }
    __syncthreads();

    // ---- fully coalesced tile store: 16 KB contiguous, float4 x4/thread --
    float* dst = out + rho0 * DD;
    #pragma unroll
    for (int i = 0; i < 4; ++i) {
        const int idx = tid + i * 256;         // float4 index within tile
        *(float4*)(dst + idx * 4) = *(const float4*)(stage + idx * 4);
    }
}

extern "C" void kernel_launch(void* const* d_in, const int* in_sizes, int n_in,
                              void* d_out, int out_size, void* d_ws, size_t ws_size,
                              hipStream_t stream) {
    const float* msa  = (const float*)d_in[0];
    const float* ln_w = (const float*)d_in[1];
    const float* ln_b = (const float*)d_in[2];
    const float* Wq   = (const float*)d_in[3];
    const float* Wk   = (const float*)d_in[4];
    const float* Wv   = (const float*)d_in[5];
    const float* Wg   = (const float*)d_in[6];
    const float* bg   = (const float*)d_in[7];
    const float* Wo   = (const float*)d_in[8];
    const float* bo   = (const float*)d_in[9];
    float* out = (float*)d_out;

    float*    xsum = (float*)d_ws;                     // 65536 f32
    float*    outw = xsum + LL * DD;                   // 65536 f32
    _Float16* kws  = (_Float16*)(outw + LL * DD);      // 4 Mi f16
    _Float16* vws  = kws + (size_t)LL * NN * 8;        // 4 Mi f16
    _Float16* bfKV = vws + (size_t)LL * NN * 8;        // 1024 f16
    _Float16* bfG  = bfKV + 1024;                      // 4096 f16
    _Float16* bfO  = bfG + 4096;                       // 4096 f16

    mcga_zero<<<(LL * DD + 255) / 256, 256, 0, stream>>>(xsum);
    mcga_prepw<<<1, 256, 0, stream>>>(Wk, Wv, Wg, Wo, bfKV, bfG, bfO);
    mcga_pass1<<<(NN / 16) * (LL / 4), 256, 0, stream>>>(msa, ln_w, ln_b, bfKV,
                                                         xsum, kws, vws);
    mcga_pass2<<<LL, 256, 0, stream>>>(xsum, Wq, kws, vws, outw);
    mcga_pass3<<<(int)(((size_t)NN * LL) / 64), 256, 0, stream>>>(
        msa, ln_w, ln_b, bfG, bg, bfO, bo, outw, out);
}
